// MultiHeadAttention_88785563943249
// MI455X (gfx1250) — compile-verified, matrix operand order checked by reading, not run on hardware
//
#include <hip/hip_runtime.h>
#include <hip/hip_bf16.h>

// ---------------------------------------------------------------------------
// MI455X (gfx1250) multi-head attention, bf16 WMMA everywhere:
//   - projections: 1 wave -> 32 tok x 64 col tile, v_wmma_f32_16x16x32_bf16
//   - attention:   1 wave -> 32 q rows, flash online softmax (32 rows = 32 lanes)
//   - bf16 intermediates in workspace: Wt_q/k/v (NxK), Q/K [B,H,S,64], V^T [B,H,64,S]
// Dims: EMB=1024, QK=V=64, HEAD=16, B=2, SQ=SKV=2048
// ---------------------------------------------------------------------------

#define EMB   1024
#define DHEAD 64
#define NHEAD 16
#define BATCH 2
#define SQL   2048
#define SKVL  2048

typedef __bf16  v16bf __attribute__((ext_vector_type(16)));
typedef float   v8f   __attribute__((ext_vector_type(8)));
typedef unsigned int u32x4 __attribute__((ext_vector_type(4)));
typedef float   f32x4 __attribute__((ext_vector_type(4)));

union Frag {
    v16bf v;
    u32x4 q[2];
    __bf16 e[16];
};

__device__ inline v8f wmma_bf16(v16bf a, v16bf b, v8f c) {
    return __builtin_amdgcn_wmma_f32_16x16x32_bf16(false, a, false, b,
                                                   (short)0, c, false, false);
}

// 16x32 bf16 row-major tile -> wave32 A/B fragment:
//   lane = (row & 15) + 16*half ; c0 = half*8
//   regs 0-3 : k = c0..c0+7 ; regs 4-7 : k = c0+16..c0+23
__device__ inline v16bf load_frag_bf16(const __bf16* base, int ld, int lane) {
    const int row = lane & 15;
    const int c0  = (lane >> 4) << 3;
    const __bf16* p = base + (size_t)row * ld + c0;
    Frag f;
    f.q[0] = *reinterpret_cast<const u32x4*>(p);
    f.q[1] = *reinterpret_cast<const u32x4*>(p + 16);
    return f.v;
}

// Same fragment from fp32 row-major source, native cvt to bf16.
__device__ inline v16bf load_frag_f32(const float* base, int ld, int lane) {
    const int row = lane & 15;
    const int c0  = (lane >> 4) << 3;
    const float* p = base + (size_t)row * ld + c0;
    f32x4 a0 = *reinterpret_cast<const f32x4*>(p);
    f32x4 a1 = *reinterpret_cast<const f32x4*>(p + 4);
    f32x4 b0 = *reinterpret_cast<const f32x4*>(p + 16);
    f32x4 b1 = *reinterpret_cast<const f32x4*>(p + 20);
    Frag f;
#pragma unroll
    for (int i = 0; i < 4; ++i) {
        f.e[i]      = (__bf16)a0[i];
        f.e[4 + i]  = (__bf16)a1[i];
        f.e[8 + i]  = (__bf16)b0[i];
        f.e[12 + i] = (__bf16)b1[i];
    }
    return f.v;
}

// ---------------------------------------------------------------------------
// W (EMB x N, fp32, row-major) -> Wt (N x EMB, bf16, row-major)
// ---------------------------------------------------------------------------
__global__ __launch_bounds__(256) void transpose_w_kernel(
    const float* __restrict__ W, __bf16* __restrict__ Wt)
{
    int idx = blockIdx.x * 256 + threadIdx.x;      // 0 .. 1024*1024-1
    int n = idx & (EMB - 1);                       // coalesced read along n
    int k = idx >> 10;
    Wt[(size_t)n * EMB + k] = (__bf16)W[(size_t)k * EMB + n];
}

// ---------------------------------------------------------------------------
// Projection: 1 wave -> 32 tokens x 64 cols; Out[b,h,s,d] bf16.
// A-frags reused 4x (n-tiles), B-frags reused 2x (q sub-tiles): 8 WMMAs/k-step
// ---------------------------------------------------------------------------
__global__ __launch_bounds__(32) void proj_qk_kernel(
    const float* __restrict__ X, const __bf16* __restrict__ Wt,
    const float* __restrict__ bias, __bf16* __restrict__ Out)
{
    const int lane = threadIdx.x;
    const int m0   = blockIdx.x * 32;              // global token tile (32 rows)
    const int h    = blockIdx.y;                   // 64-col block == head
    const int bb   = m0 >> 11;                     // /2048
    const int s0   = m0 & (SQL - 1);

    v8f acc[2][4] = {};
    for (int k0 = 0; k0 < EMB; k0 += 32) {
        v16bf a0 = load_frag_f32(X + (size_t)m0 * EMB + k0, EMB, lane);
        v16bf a1 = load_frag_f32(X + (size_t)(m0 + 16) * EMB + k0, EMB, lane);
#pragma unroll
        for (int t = 0; t < 4; ++t) {
            v16bf bfrg = load_frag_bf16(
                Wt + (size_t)(h * DHEAD + t * 16) * EMB + k0, EMB, lane);
            acc[0][t] = wmma_bf16(a0, bfrg, acc[0][t]);
            acc[1][t] = wmma_bf16(a1, bfrg, acc[1][t]);
        }
    }
    const int n = lane & 15, half = lane >> 4;
    __bf16* op = Out + (size_t)(bb * NHEAD + h) * SQL * DHEAD;
#pragma unroll
    for (int t = 0; t < 4; ++t) {
        float bv = bias[h * DHEAD + t * 16 + n];
#pragma unroll
        for (int qt = 0; qt < 2; ++qt)
#pragma unroll
            for (int r = 0; r < 8; ++r) {
                int s = s0 + qt * 16 + half * 8 + r;
                op[(size_t)s * DHEAD + t * 16 + n] = (__bf16)(acc[qt][t][r] + bv);
            }
    }
}

// Projection writing V transposed: Vt[b,h,d,skv] bf16 (contiguous skv -> 16B stores)
__global__ __launch_bounds__(32) void proj_v_kernel(
    const float* __restrict__ X, const __bf16* __restrict__ Wt,
    const float* __restrict__ bias, __bf16* __restrict__ Vt)
{
    const int lane = threadIdx.x;
    const int m0   = blockIdx.x * 32;
    const int h    = blockIdx.y;
    const int bb   = m0 >> 11;
    const int s0   = m0 & (SKVL - 1);

    v8f acc[2][4] = {};
    for (int k0 = 0; k0 < EMB; k0 += 32) {
        v16bf a0 = load_frag_f32(X + (size_t)m0 * EMB + k0, EMB, lane);
        v16bf a1 = load_frag_f32(X + (size_t)(m0 + 16) * EMB + k0, EMB, lane);
#pragma unroll
        for (int t = 0; t < 4; ++t) {
            v16bf bfrg = load_frag_bf16(
                Wt + (size_t)(h * DHEAD + t * 16) * EMB + k0, EMB, lane);
            acc[0][t] = wmma_bf16(a0, bfrg, acc[0][t]);
            acc[1][t] = wmma_bf16(a1, bfrg, acc[1][t]);
        }
    }
    const int n = lane & 15, half = lane >> 4;
#pragma unroll
    for (int t = 0; t < 4; ++t) {
        float bv = bias[h * DHEAD + t * 16 + n];
        __bf16* row = Vt + ((size_t)(bb * NHEAD + h) * DHEAD + (t * 16 + n)) * SKVL;
#pragma unroll
        for (int qt = 0; qt < 2; ++qt) {
            union { u32x4 q; __bf16 e[8]; } pk;
#pragma unroll
            for (int r = 0; r < 8; ++r) pk.e[r] = (__bf16)(acc[qt][t][r] + bv);
            *reinterpret_cast<u32x4*>(row + s0 + qt * 16 + half * 8) = pk.q;
        }
    }
}

// ---------------------------------------------------------------------------
// Flash attention: 1 wave per (b, h, 32-query-row tile), kv in steps of 32.
// Every K/V fragment feeds 2 WMMAs (two q sub-tiles); softmax is divergence-free
// (32 rows <-> 32 lanes).
// ---------------------------------------------------------------------------
__global__ __launch_bounds__(32) void attn_kernel(
    const __bf16* __restrict__ Qb, const __bf16* __restrict__ Kb,
    const __bf16* __restrict__ Vt, const unsigned char* __restrict__ mask,
    float* __restrict__ out)
{
    const int lane = threadIdx.x;
    const int q0 = blockIdx.x * 32;
    const int h  = blockIdx.y;
    const int b  = blockIdx.z;

    const __bf16* Qbase = Qb + ((size_t)(b * NHEAD + h) * SQL + q0) * DHEAD;
    const __bf16* Kbase = Kb + (size_t)(b * NHEAD + h) * SKVL * DHEAD;
    const __bf16* Vbase = Vt + (size_t)(b * NHEAD + h) * DHEAD * SKVL;
    const unsigned char* Mbase = mask + ((size_t)b * SQL + q0) * SKVL;

    // Q fragments: 2 q sub-tiles x 2 k-halves (d=64), reused for all kv steps
    v16bf qf[2][2];
#pragma unroll
    for (int qt = 0; qt < 2; ++qt) {
        qf[qt][0] = load_frag_bf16(Qbase + (size_t)qt * 16 * DHEAD, DHEAD, lane);
        qf[qt][1] = load_frag_bf16(Qbase + (size_t)qt * 16 * DHEAD + 32, DHEAD, lane);
    }

    __shared__ float  sS[32][33];                    // scores, padded rows
    __shared__ __align__(16) __bf16 sP[32][32];      // probabilities
    __shared__ float  sAlpha[32];
    __shared__ float  sL[32];

    v8f o[2][4] = {};
    float run_m = -3.0e38f, run_l = 0.0f;            // lane owns query row `lane`
    const int nn = lane & 15, half = lane >> 4;
    const float scale = 0.125f;                      // 1/sqrt(64)

    for (int kv0 = 0; kv0 < SKVL; kv0 += 32) {
        if (kv0 + 32 < SKVL) {                       // prefetch next kv slab
            __builtin_prefetch(Kbase + (size_t)(kv0 + 32) * DHEAD, 0, 1);
            __builtin_prefetch(Vbase + (size_t)kv0 + 32, 0, 1);
        }
        // ---- S = Q K^T for 32x32 slab (8 WMMAs) -> LDS
#pragma unroll
        for (int t = 0; t < 2; ++t) {
            const __bf16* kp = Kbase + (size_t)(kv0 + t * 16) * DHEAD;
            v16bf kf0 = load_frag_bf16(kp, DHEAD, lane);
            v16bf kf1 = load_frag_bf16(kp + 32, DHEAD, lane);
#pragma unroll
            for (int qt = 0; qt < 2; ++qt) {
                v8f s = {};
                s = wmma_bf16(qf[qt][0], kf0, s);
                s = wmma_bf16(qf[qt][1], kf1, s);
#pragma unroll
                for (int r = 0; r < 8; ++r)
                    sS[qt * 16 + half * 8 + r][t * 16 + nn] = s[r];
            }
        }
        __syncthreads();

        // ---- online softmax: lane owns query row `lane` of the 32-row tile
        {
            const unsigned char* mrow = Mbase + (size_t)lane * SKVL + kv0;
            union { u32x4 v[2]; unsigned char bch[32]; } mb;
            mb.v[0] = *reinterpret_cast<const u32x4*>(mrow);
            mb.v[1] = *reinterpret_cast<const u32x4*>(mrow + 16);
            float vals[32];
            float mloc = -3.0e38f;
#pragma unroll
            for (int j = 0; j < 32; ++j) {
                float v = sS[lane][j] * scale;
                v = mb.bch[j] ? -1.0e9f : v;         // mask==true -> -1e9
                vals[j] = v;
                mloc = fmaxf(mloc, v);
            }
            float newm  = fmaxf(run_m, mloc);
            float alpha = __expf(run_m - newm);
            float lsum  = 0.0f;
#pragma unroll
            for (int j = 0; j < 32; ++j) {
                float p = __expf(vals[j] - newm);
                lsum += p;
                sP[lane][j] = (__bf16)p;
            }
            run_l = run_l * alpha + lsum;
            run_m = newm;
            sAlpha[lane] = alpha;
        }
        __syncthreads();

        // ---- O = alpha*O + P V   (8 WMMAs; each V fragment feeds 2)
        v16bf pf[2];
        pf[0] = load_frag_bf16(&sP[0][0], 32, lane);
        pf[1] = load_frag_bf16(&sP[16][0], 32, lane);
        float al[2][8];
#pragma unroll
        for (int qt = 0; qt < 2; ++qt)
#pragma unroll
            for (int r = 0; r < 8; ++r) al[qt][r] = sAlpha[qt * 16 + half * 8 + r];
#pragma unroll
        for (int t = 0; t < 4; ++t) {
            v16bf vf = load_frag_bf16(Vbase + (size_t)(t * 16) * SKVL + kv0,
                                      SKVL, lane);
#pragma unroll
            for (int qt = 0; qt < 2; ++qt) {
#pragma unroll
                for (int r = 0; r < 8; ++r) o[qt][t][r] *= al[qt][r];
                o[qt][t] = wmma_bf16(pf[qt], vf, o[qt][t]);
            }
        }
        __syncthreads();
    }

    // ---- normalize by l, store fp32 output [B, SQ, H*64]
    sL[lane] = run_l;
    __syncthreads();
    float* obase = out + ((size_t)b * SQL + q0) * (NHEAD * DHEAD) + h * DHEAD;
#pragma unroll
    for (int qt = 0; qt < 2; ++qt) {
        float linv[8];
#pragma unroll
        for (int r = 0; r < 8; ++r)
            linv[r] = 1.0f / fmaxf(sL[qt * 16 + half * 8 + r], 1e-30f);
#pragma unroll
        for (int t = 0; t < 4; ++t)
#pragma unroll
            for (int r = 0; r < 8; ++r)
                obase[(size_t)(qt * 16 + half * 8 + r) * (NHEAD * DHEAD) + t * 16 + nn] =
                    o[qt][t][r] * linv[r];
    }
}

// ---------------------------------------------------------------------------
extern "C" void kernel_launch(void* const* d_in, const int* in_sizes, int n_in,
                              void* d_out, int out_size, void* d_ws, size_t ws_size,
                              hipStream_t stream) {
    const float*         x_q  = (const float*)d_in[0];
    const float*         x_kv = (const float*)d_in[1];
    const unsigned char* mask = (const unsigned char*)d_in[2];
    const float*         Wq   = (const float*)d_in[3];
    const float*         bq   = (const float*)d_in[4];
    const float*         Wk   = (const float*)d_in[5];
    const float*         bk   = (const float*)d_in[6];
    const float*         Wv   = (const float*)d_in[7];
    const float*         bv   = (const float*)d_in[8];
    float*               out  = (float*)d_out;

    char* ws = (char*)d_ws;
    __bf16* Wtq = (__bf16*)(ws);                        // 2 MB
    __bf16* Wtk = (__bf16*)(ws + (size_t)2  * 1048576); // 2 MB
    __bf16* Wtv = (__bf16*)(ws + (size_t)4  * 1048576); // 2 MB
    __bf16* Qb  = (__bf16*)(ws + (size_t)6  * 1048576); // 8 MB
    __bf16* Kb  = (__bf16*)(ws + (size_t)14 * 1048576); // 8 MB
    __bf16* Vt  = (__bf16*)(ws + (size_t)22 * 1048576); // 8 MB

    dim3 tt(256), tg((EMB * EMB) / 256);
    transpose_w_kernel<<<tg, tt, 0, stream>>>(Wq, Wtq);
    transpose_w_kernel<<<tg, tt, 0, stream>>>(Wk, Wtk);
    transpose_w_kernel<<<tg, tt, 0, stream>>>(Wv, Wtv);

    const int tok_tiles = (BATCH * SQL) / 32;           // 128
    dim3 pb(32);
    proj_qk_kernel<<<dim3(tok_tiles, NHEAD), pb, 0, stream>>>(x_q,  Wtq, bq, Qb);
    proj_qk_kernel<<<dim3(tok_tiles, NHEAD), pb, 0, stream>>>(x_kv, Wtk, bk, Kb);
    proj_v_kernel <<<dim3(tok_tiles, NHEAD), pb, 0, stream>>>(x_kv, Wtv, bv, Vt);

    attn_kernel<<<dim3(SQL / 32, NHEAD, BATCH), pb, 0, stream>>>(Qb, Kb, Vt, mask, out);
}
